// Net_AGNN_11914239279709
// MI455X (gfx1250) — compile-verified, matrix operand order checked by reading, not run on hardware
//
#include <hip/hip_runtime.h>
#include <hip/hip_bf16.h>
#include <math.h>

// ---------- types for WMMA / TDM ----------
typedef __attribute__((ext_vector_type(16))) __bf16 v16bf;
typedef __attribute__((ext_vector_type(8)))  float  v8f;
typedef __attribute__((ext_vector_type(4)))  unsigned int v4u;
typedef __attribute__((ext_vector_type(8)))  int    v8i;
typedef __attribute__((ext_vector_type(4)))  int    v4i;

static __device__ __forceinline__ __bf16 f2bf(float f) {
    return (__bf16)f;   // v_cvt_pk_bf16_f32 when paired
}

// =====================================================================
// Kernel 1: h = relu(x[N,512] @ W1[512,32] + b1)  via v_wmma_f32_16x16x32_bf16
// Block = 256 threads = 8 waves; each wave computes one 16x16 output tile;
// block covers 64 rows x 32 cols.
// W1 (64 KB fp32) is DMA'd into LDS by the Tensor Data Mover, then repacked
// once into a fragment-major bf16 layout (32 KB): one contiguous v16bf per
// (k-step, column, lane-half) so the k-loop does 2x ds_load_b128 per step.
// =====================================================================
__global__ void gemm_relu_wmma(const float* __restrict__ x,
                               const float* __restrict__ W1,
                               const float* __restrict__ b1,
                               float* __restrict__ h, int N) {
    __shared__ float sWf[512 * 32];     // 64 KB, TDM destination (fp32 W1)
    __shared__ v16bf sWB[16 * 32 * 2];  // 32 KB, fragment-major bf16 B operands

    // --- TDM: wave 0 issues one tensor DMA of all of W1 into LDS ---
    if (threadIdx.x < 32) {
        unsigned int       lds = (unsigned int)(uintptr_t)&sWf[0]; // LDS aperture low bits
        unsigned long long ga  = (unsigned long long)(uintptr_t)W1;
        const unsigned int NE  = 512u * 32u;                       // 16384 fp32 elements
        v4u g0;
        g0[0] = 1u;                                        // count=1, user descriptor
        g0[1] = lds;                                       // lds_addr
        g0[2] = (unsigned int)(ga & 0xFFFFFFFFu);          // global_addr[31:0]
        g0[3] = (2u << 30) | ((unsigned int)(ga >> 32) & 0x01FFFFFFu); // type=2 | ga[56:32]
        v8i g1;
        g1[0] = (int)(2u << 16);                           // wg_mask=0, data_size=4B
        g1[1] = (int)((NE & 0xFFFFu) << 16);               // tensor_dim0[15:0]
        g1[2] = (int)(((NE >> 16) & 0xFFFFu) | (1u << 16));// tensor_dim0[31:16] | tensor_dim1=1
        g1[3] = (int)((NE & 0xFFFFu) << 16);               // tile_dim0 = 16384
        g1[4] = 0;                                         // tile_dim1/2 unused
        g1[5] = (int)NE;                                   // tensor_dim0_stride[31:0]
        g1[6] = 0;
        g1[7] = 0;
        v4i g2 = {0, 0, 0, 0};
        v4i g3 = {0, 0, 0, 0};
        v8i g4 = {0, 0, 0, 0, 0, 0, 0, 0};
        __builtin_amdgcn_tensor_load_to_lds(g0, g1, g2, g3, g4, 0);
        __builtin_amdgcn_s_wait_tensorcnt(0);
    }
    __syncthreads();

    // One-time repack fp32 -> fragment-major bf16.
    // Fragment j = kstep*64 + col*2 + half holds the 16 K-values that
    // lane (col,half) feeds to the WMMA B operand at k0 = kstep*32.
    for (int j = threadIdx.x; j < 16 * 32 * 2; j += 256) {
        const int kstep = j >> 6;
        const int col   = (j & 63) >> 1;
        const int hf    = j & 1;
        const int kb    = kstep * 32 + hf * 8;
        v16bf f;
#pragma unroll
        for (int i = 0; i < 8; ++i) {
            f[i]     = f2bf(sWf[(kb + i)      * 32 + col]);
            f[i + 8] = f2bf(sWf[(kb + i + 16) * 32 + col]);
        }
        sWB[j] = f;
    }
    __syncthreads();

    const int wave = threadIdx.x >> 5;
    const int lane = threadIdx.x & 31;
    const int rowTile = blockIdx.x * 4 + (wave >> 1);
    const int colTile = wave & 1;
    const int row0 = rowTile * 16;
    if (row0 >= N) return;
    const int col0 = colTile * 16;

    const int half = lane >> 4;     // lanes 0-15 vs 16-31
    const int l    = lane & 15;
    int arow = row0 + l; if (arow >= N) arow = N - 1;

    const int fragBase = (col0 + l) * 2 + half;   // + kstep*64 per step

    v8f acc = {};
#pragma unroll 2
    for (int k0 = 0; k0 < 512; k0 += 32) {
        // A (16x32 bf16): half 0 holds K = {k0+0..7, k0+16..23},
        //                 half 1 holds K = {k0+8..15, k0+24..31}
        v16bf a;
        const float* xp = x + (size_t)arow * 512 + k0 + half * 8;
#pragma unroll
        for (int i = 0; i < 8; ++i) {
            a[i]     = f2bf(xp[i]);
            a[i + 8] = f2bf(xp[i + 16]);
        }
        v16bf b = sWB[(k0 >> 5) * 64 + fragBase];   // 2x ds_load_b128
        acc = __builtin_amdgcn_wmma_f32_16x16x32_bf16(
            false, a, false, b, (short)0, acc, false, false);
    }

    // C/D layout: VGPR r -> M = r + 8*half, N = l
    const float bias = b1[col0 + l];
    float* hp = h + (size_t)(row0 + half * 8) * 32 + (col0 + l);
    if (row0 + 16 <= N) {           // full-tile fast path (all but last tile)
#pragma unroll
        for (int r = 0; r < 8; ++r)
            hp[(size_t)r * 32] = fmaxf(acc[r] + bias, 0.0f);
    } else {
#pragma unroll
        for (int r = 0; r < 8; ++r)
            if (row0 + r + half * 8 < N)
                hp[(size_t)r * 32] = fmaxf(acc[r] + bias, 0.0f);
    }
}

// =====================================================================
// Kernel 2: per-layer prologue. One wave per row (lane = feature):
//   hn = h / max(||h||, eps); zero softmax denominator s and h_next.
// =====================================================================
__global__ void row_norm_zero(const float* __restrict__ h,
                              float* __restrict__ hn,
                              float* __restrict__ s,
                              float* __restrict__ hnext, int N) {
    int row  = blockIdx.x * 8 + (threadIdx.x >> 5);
    int lane = threadIdx.x & 31;
    if (row >= N) return;
    float v  = h[(size_t)row * 32 + lane];
    float sq = v * v;
#pragma unroll
    for (int off = 16; off > 0; off >>= 1) sq += __shfl_xor(sq, off, 32);
    float inv = 1.0f / fmaxf(sqrtf(sq), 1e-12f);
    hn[(size_t)row * 32 + lane]    = v * inv;
    hnext[(size_t)row * 32 + lane] = 0.0f;
    if (lane == 0) s[row] = 0.0f;
}

// =====================================================================
// Kernel 3: edge pass 1. One edge per wave; coalesced 128B row gathers
// (hn lives in the 192MB L2). alpha = <hn[src],hn[dst]> in [-1,1], so the
// segment_max stabilization pass is skipped (exp cannot overflow).
// Virtual edges e >= E are the self-loops (i,i).
// =====================================================================
__global__ void edge_alpha(const float* __restrict__ hn,
                           const long long* __restrict__ ei,
                           float* __restrict__ ew,
                           float* __restrict__ s,
                           long long E, long long N) {
    const long long total  = E + N;
    const int  lane   = threadIdx.x & 31;
    long long  wid    = (long long)blockIdx.x * 8 + (threadIdx.x >> 5);
    const long long stride = (long long)gridDim.x * 8;
    for (long long e = wid; e < total; e += stride) {
        long long en = e + stride;                 // prefetch next index pair
        if (en < E) {
            __builtin_prefetch(&ei[en], 0, 0);
            __builtin_prefetch(&ei[E + en], 0, 0);
        }
        long long srcI, dstI;
        if (e < E) { srcI = ei[e]; dstI = ei[E + e]; }
        else       { srcI = dstI = e - E; }
        float a = hn[srcI * 32 + lane] * hn[dstI * 32 + lane];
#pragma unroll
        for (int off = 16; off > 0; off >>= 1) a += __shfl_xor(a, off, 32);
        float ex = __expf(a);
        if (lane == 0) {
            ew[e] = ex;
            atomicAdd(&s[dstI], ex);
        }
    }
}

// =====================================================================
// Kernel 4: edge pass 2. coef = e / s[dst]; h_next[dst] += coef * h[src].
// One edge per wave; one coalesced global_atomic_add_f32 row per edge.
// =====================================================================
__global__ void edge_aggregate(const float* __restrict__ h,
                               const long long* __restrict__ ei,
                               const float* __restrict__ ew,
                               const float* __restrict__ s,
                               float* __restrict__ hnext,
                               long long E, long long N) {
    const long long total  = E + N;
    const int  lane   = threadIdx.x & 31;
    long long  wid    = (long long)blockIdx.x * 8 + (threadIdx.x >> 5);
    const long long stride = (long long)gridDim.x * 8;
    for (long long e = wid; e < total; e += stride) {
        long long en = e + stride;
        if (en < E) {
            __builtin_prefetch(&ei[en], 0, 0);
            __builtin_prefetch(&ei[E + en], 0, 0);
        }
        long long srcI, dstI;
        if (e < E) { srcI = ei[e]; dstI = ei[E + e]; }
        else       { srcI = dstI = e - E; }
        float c0 = 0.0f;
        if (lane == 0) c0 = ew[e] / s[dstI];       // single scalar load+div
        float coef = __shfl(c0, 0, 32);            // broadcast to the wave
        float v    = h[srcI * 32 + lane] * coef;
        atomicAdd(&hnext[dstI * 32 + lane], v);
    }
}

// =====================================================================
// Kernel 5: logits = h[N,32] @ W2[32,40] + b2, then log_softmax.
// Thread per row; W2/b2 staged in LDS.
// =====================================================================
__global__ void head_logsoftmax(const float* __restrict__ h,
                                const float* __restrict__ W2,
                                const float* __restrict__ b2,
                                float* __restrict__ out, int N) {
    __shared__ float sW[32 * 40];
    __shared__ float sb[40];
    for (int i = threadIdx.x; i < 32 * 40; i += blockDim.x) sW[i] = W2[i];
    for (int i = threadIdx.x; i < 40;      i += blockDim.x) sb[i] = b2[i];
    __syncthreads();
    int row = blockIdx.x * blockDim.x + threadIdx.x;
    if (row >= N) return;

    float hr[32];
    const float4* hp = (const float4*)(h + (size_t)row * 32);
#pragma unroll
    for (int i = 0; i < 8; ++i) {
        float4 v = hp[i];
        hr[4*i+0] = v.x; hr[4*i+1] = v.y; hr[4*i+2] = v.z; hr[4*i+3] = v.w;
    }
    float lg[40];
    float m = -INFINITY;
#pragma unroll
    for (int c = 0; c < 40; ++c) {
        float acc = sb[c];
#pragma unroll
        for (int k = 0; k < 32; ++k) acc = fmaf(hr[k], sW[k * 40 + c], acc);
        lg[c] = acc;
        m = fmaxf(m, acc);
    }
    float sum = 0.0f;
#pragma unroll
    for (int c = 0; c < 40; ++c) sum += __expf(lg[c] - m);
    float lse = m + __logf(sum);
    float* op = out + (size_t)row * 40;
#pragma unroll
    for (int c = 0; c < 40; ++c) op[c] = lg[c] - lse;
}

// =====================================================================
// Host-side orchestration (graph-capture safe: launches only).
// Workspace layout (floats): hA[N*32] | hB[N*32] | hn[N*32] | s[N] | ew[E+N]
// ~52 MB total.
// =====================================================================
extern "C" void kernel_launch(void* const* d_in, const int* in_sizes, int n_in,
                              void* d_out, int out_size, void* d_ws, size_t ws_size,
                              hipStream_t stream) {
    const float*     x  = (const float*)d_in[0];
    const long long* ei = (const long long*)d_in[1];   // int64 edge_index [2,E]
    const float*     W1 = (const float*)d_in[2];
    const float*     b1 = (const float*)d_in[3];
    const float*     W2 = (const float*)d_in[4];
    const float*     b2 = (const float*)d_in[5];

    const long long N = in_sizes[0] / 512;
    const long long E = in_sizes[1] / 2;

    float* hA = (float*)d_ws;
    float* hB = hA + N * 32;
    float* hn = hB + N * 32;
    float* s  = hn + N * 32;
    float* ew = s + N;

    // 1) Feature transform with WMMA (+ TDM weight staging).
    int gemmBlocks = (int)((N + 63) / 64);
    gemm_relu_wmma<<<gemmBlocks, 256, 0, stream>>>(x, W1, b1, hA, (int)N);

    // 2) 32 AGNN layers.
    const long long totalEdges = E + N;
    int waveBlocks = (int)((totalEdges + 7) / 8);   // 1 edge per wave, 8 waves/block
    int rnBlocks   = (int)((N + 7) / 8);
    float* cur = hA;
    float* nxt = hB;
    for (int layer = 0; layer < 32; ++layer) {
        row_norm_zero <<<rnBlocks,   256, 0, stream>>>(cur, hn, s, nxt, (int)N);
        edge_alpha    <<<waveBlocks, 256, 0, stream>>>(hn, ei, ew, s, E, N);
        edge_aggregate<<<waveBlocks, 256, 0, stream>>>(cur, ei, ew, s, nxt, E, N);
        float* t = cur; cur = nxt; nxt = t;
    }

    // 3) Classifier head + log_softmax.
    int headBlocks = (int)((N + 255) / 256);
    head_logsoftmax<<<headBlocks, 256, 0, stream>>>(cur, W2, b2, (float*)d_out, (int)N);
}
